// Model_13348758356240
// MI455X (gfx1250) — matrix operand order repaired and checked by reading, not compiled, not run on hardware
//
#include <hip/hip_runtime.h>

typedef long long ll;

typedef __attribute__((ext_vector_type(16))) __bf16 bf16x16;
typedef __attribute__((ext_vector_type(8)))  __bf16 bf16x8;
typedef __attribute__((ext_vector_type(8)))  float  f32x8;
typedef int v4i __attribute__((vector_size(16)));
typedef int v8i __attribute__((vector_size(32)));

static constexpr int B_ = 128, L_ = 49, D_ = 2048, H_ = 2048, E_ = 512, K_ = 512, V_ = 10000, T_ = 20;
static constexpr int GATES = 4 * H_;        // 8192
static constexpr int XED   = E_ + D_;       // 2560
static constexpr int KCAT  = E_ + D_ + H_;  // 4608

__device__ __forceinline__ __bf16 f2bf(float x) {
    unsigned u = __builtin_bit_cast(unsigned, x);
    u += 0x7fffu + ((u >> 16) & 1u);           // round-to-nearest-even
    unsigned short s = (unsigned short)(u >> 16);
    return __builtin_bit_cast(__bf16, s);
}
__device__ __forceinline__ float sigm(float x) { return 1.0f / (1.0f + expf(-x)); }

// ---- CDNA5 async global->LDS path (ASYNCcnt), with fallback ---------------
#if defined(__has_builtin)
#if __has_builtin(__builtin_amdgcn_global_load_async_to_lds_b128) && \
    __has_builtin(__builtin_amdgcn_s_wait_asynccnt)
#define HAVE_ASYNC_LDS 1
#endif
#if defined(HAVE_ASYNC_LDS) && \
    __has_builtin(__builtin_amdgcn_tensor_load_to_lds) && \
    __has_builtin(__builtin_amdgcn_s_wait_tensorcnt)
#define HAVE_TDM 1
#endif
#endif

#ifdef HAVE_ASYNC_LDS
__device__ __forceinline__ void async_cp16(const __bf16* g, __bf16* l) {
    __builtin_amdgcn_global_load_async_to_lds_b128(
        (__attribute__((address_space(1))) v4i*)(v4i*)g,
        (__attribute__((address_space(3))) v4i*)(v4i*)l, 0, 0);
}
#endif

#ifdef HAVE_TDM
// TDM descriptor-driven DMA of a [rows x 32] bf16 tile (row stride ldw elems)
// into LDS at byte offset lds_off, padded +16B after each 64B row -> stride 40
// halfs. Rows beyond `rows` (tensor_dim1) are zero-filled by OOB semantics.
__device__ __forceinline__ void tdm_load_w(const __bf16* gsrc, unsigned lds_off,
                                           unsigned rows, unsigned ldw_elems) {
    unsigned long long ga = (unsigned long long)gsrc;
    // D# group 0: count=1 | lds_addr | global_addr[56:0] | type=2
    v4i g0 = { 1,
               (int)lds_off,
               (int)(unsigned)(ga & 0xffffffffu),
               (int)(((unsigned)(ga >> 32) & 0x01ffffffu) | (2u << 30)) };
    // D# group 1
    unsigned d0 = (1u << 16)            // data_size = 2 bytes
                | (1u << 20)            // pad_enable
                | (3u << 22)            // pad_interval: 16 DWORDs (= one 64B row)
                | (3u << 25);           // pad_amount: 4 DWORDs (= 16B)
    unsigned d1 = (32u & 0xffffu) << 16;                       // tensor_dim0 = 32 (lo)
    unsigned d2 = 0u | ((rows & 0xffffu) << 16);               // td0 hi | tensor_dim1 lo
    unsigned d3 = ((rows >> 16) & 0xffffu) | (32u << 16);      // td1 hi | tile_dim0 = 32
    unsigned d4 = 64u;                                         // tile_dim1 = 64, tile_dim2 = 0
    unsigned d5 = ldw_elems;                                   // tensor_dim0_stride lo32
    v8i g1 = { (int)d0, (int)d1, (int)d2, (int)d3, (int)d4, (int)d5, 0, 0 };
    v4i z4 = { 0, 0, 0, 0 };
    v8i z8 = { 0, 0, 0, 0, 0, 0, 0, 0 };
    __builtin_amdgcn_tensor_load_to_lds(g0, g1, z4, z4, z8, 0);
}
#endif

// ---------------------------------------------------------------------------
// bf16 WMMA GEMM:  C[m,n] = sum_k A[m,k] * W[n,k] (+ bias[n])
// A: [M, lda] bf16 row-major, W: [N, ldw] bf16 row-major (i.e. B = W^T).
// Block tile 128x64, 8 waves (4 M-waves x 2 N-waves), each wave 2x2 WMMA tiles.
// Double-buffered LDS. A tile: GLOBAL_LOAD_ASYNC_TO_LDS_B128 (ASYNCcnt).
// W tile: TENSOR_LOAD_TO_LDS via TDM, issued by wave 0 (TENSORcnt).
// grid = (ceil(N/64), M/128)
// ---------------------------------------------------------------------------
#define LDS_STRIDE 40   // 32 halfs + 8 pad: 80B rows -> conflict-free, 16B aligned

union AFrag { bf16x16 v; bf16x8 h[2]; };

__global__ __launch_bounds__(256) void gemm_bf16_wt(
    const __bf16* __restrict__ A, ll lda,
    const __bf16* __restrict__ W, ll ldw,
    const float* __restrict__ bias,
    float* __restrict__ C, ll ldc,
    int N, int Kdim)
{
    __shared__ __bf16 sA[2][128 * LDS_STRIDE];
    __shared__ __bf16 sW[2][64 * LDS_STRIDE];

    const int tid  = threadIdx.x;
    const int lane = tid & 31;
    const int wid  = tid >> 5;
    const int wm   = wid & 3;      // 0..3 -> 32-row strip
    const int wn   = wid >> 2;     // 0..1 -> 32-col strip
    const int hf   = lane >> 4;    // lane half (0/1)
    const int lm   = lane & 15;

    const int nb = blockIdx.x * 64;
    const ll  mb = (ll)blockIdx.y * 128;

    // per-thread A staging coordinates (2 x 16B per thread)
    const int ar0 = tid >> 2,          ac0 = (tid & 3) * 8;
    const int ar1 = (tid + 256) >> 2,  ac1 = ((tid + 256) & 3) * 8;
    // W staging (fallback path): branch-free clamped row
    int wr = nb + (tid >> 2); if (wr > N - 1) wr = N - 1;
    const int wc = (tid & 3) * 8;
    const int wlr = tid >> 2;
    const unsigned wrows = (unsigned)((N - nb) < 64 ? (N - nb) : 64);

    f32x8 acc[2][2];
    acc[0][0] = {}; acc[0][1] = {}; acc[1][0] = {}; acc[1][1] = {};

    const int nk = Kdim >> 5;   // K tiles of 32 (all K here are multiples of 32)

    // ---- stage one K-tile into LDS buffer `buf` ----
    auto stage = [&](int buf, int k0) {
#ifdef HAVE_ASYNC_LDS
        async_cp16(A + (mb + ar0) * lda + k0 + ac0, &sA[buf][ar0 * LDS_STRIDE + ac0]);
        async_cp16(A + (mb + ar1) * lda + k0 + ac1, &sA[buf][ar1 * LDS_STRIDE + ac1]);
#else
        bf16x8 va0 = *(const bf16x8*)(A + (mb + ar0) * lda + k0 + ac0);
        bf16x8 va1 = *(const bf16x8*)(A + (mb + ar1) * lda + k0 + ac1);
        *(bf16x8*)(&sA[buf][ar0 * LDS_STRIDE + ac0]) = va0;
        *(bf16x8*)(&sA[buf][ar1 * LDS_STRIDE + ac1]) = va1;
#endif
#ifdef HAVE_TDM
        if (wid == 0) {   // one TDM descriptor per block moves the whole W tile
            unsigned lds_off = (unsigned)(unsigned long long)
                (__attribute__((address_space(3))) __bf16*)(&sW[buf][0]);
            tdm_load_w(W + (ll)nb * ldw + k0, lds_off, wrows, (unsigned)ldw);
        }
#elif defined(HAVE_ASYNC_LDS)
        async_cp16(W + (ll)wr * ldw + k0 + wc, &sW[buf][wlr * LDS_STRIDE + wc]);
#else
        bf16x8 vw = *(const bf16x8*)(W + (ll)wr * ldw + k0 + wc);
        *(bf16x8*)(&sW[buf][wlr * LDS_STRIDE + wc]) = vw;
#endif
    };

    stage(0, 0);

    for (int ki = 0; ki < nk; ++ki) {
        const int buf = ki & 1;
        const bool more = (ki + 1) < nk;
        if (more) stage(buf ^ 1, (ki + 1) << 5);   // overlap next tile with WMMA
#ifdef HAVE_ASYNC_LDS
#ifdef HAVE_TDM
        if (more) __builtin_amdgcn_s_wait_asynccnt(2);   // A copies of current buf landed
        else      __builtin_amdgcn_s_wait_asynccnt(0);
        if (wid == 0) {
            if (more) __builtin_amdgcn_s_wait_tensorcnt(1);  // W DMA of current buf done
            else      __builtin_amdgcn_s_wait_tensorcnt(0);
        }
#else
        if (more) __builtin_amdgcn_s_wait_asynccnt(3);
        else      __builtin_amdgcn_s_wait_asynccnt(0);
#endif
#endif
        __syncthreads();

        // ---- fragments (ISA 7.12.2 layouts) ----
        AFrag a[2], b[2];
        #pragma unroll
        for (int s = 0; s < 2; ++s) {
            const __bf16* ap = &sA[buf][(wm * 32 + s * 16 + lm) * LDS_STRIDE];
            a[s].h[0] = *(const bf16x8*)(ap + hf * 8);        // K = off..off+7
            a[s].h[1] = *(const bf16x8*)(ap + 16 + hf * 8);   // K = 16+off..
            const __bf16* wp = &sW[buf][(wn * 32 + s * 16 + lm) * LDS_STRIDE];
            b[s].h[0] = *(const bf16x8*)(wp + hf * 8);        // K = off..off+7
            b[s].h[1] = *(const bf16x8*)(wp + 16 + hf * 8);
        }
        #pragma unroll
        for (int i = 0; i < 2; ++i)
            #pragma unroll
            for (int j = 0; j < 2; ++j)
                acc[i][j] = __builtin_amdgcn_wmma_f32_16x16x32_bf16(
                    false, a[i].v, false, b[j].v, (short)0, acc[i][j], false, false);
        __syncthreads();   // reads done before this buffer is overwritten
    }

    // ---- store: VGPR j -> row j + 8*hf, col = lm ----
    #pragma unroll
    for (int i = 0; i < 2; ++i) {
        #pragma unroll
        for (int j = 0; j < 2; ++j) {
            int col = nb + wn * 32 + j * 16 + lm;
            if (col < N) {
                float bv = bias ? bias[col] : 0.0f;
                #pragma unroll
                for (int r = 0; r < 8; ++r) {
                    ll row = mb + wm * 32 + i * 16 + r + hf * 8;
                    C[row * ldc + col] = acc[i][j][r] + bv;
                }
            }
        }
    }
}

// ---------------------------------------------------------------------------
// one-time conversion / packing kernels
// ---------------------------------------------------------------------------
__global__ void k_f32_to_bf16(const float* __restrict__ in, __bf16* __restrict__ out, ll n) {
    for (ll i = (ll)blockIdx.x * blockDim.x + threadIdx.x; i < n; i += (ll)gridDim.x * blockDim.x)
        out[i] = f2bf(in[i]);
}

__global__ void k_transpose_bf16(const float* __restrict__ in, __bf16* __restrict__ out, int R, int Ccols) {
    ll n = (ll)R * Ccols;
    for (ll i = (ll)blockIdx.x * blockDim.x + threadIdx.x; i < n; i += (ll)gridDim.x * blockDim.x) {
        int r = (int)(i / Ccols), c = (int)(i % Ccols);
        out[(ll)c * R + r] = f2bf(in[i]);
    }
}

__global__ void k_build_wcat(const float* __restrict__ Wih, const float* __restrict__ Whh,
                             __bf16* __restrict__ Wcat) {
    ll n = (ll)GATES * KCAT;
    for (ll i = (ll)blockIdx.x * blockDim.x + threadIdx.x; i < n; i += (ll)gridDim.x * blockDim.x) {
        int row = (int)(i / KCAT), k = (int)(i % KCAT);
        float v = (k < XED) ? Wih[(ll)row * XED + k] : Whh[(ll)row * H_ + (k - XED)];
        Wcat[i] = f2bf(v);
    }
}

__global__ void k_bias_cat(const float* a, const float* b, float* c) {
    int i = blockIdx.x * blockDim.x + threadIdx.x;
    if (i < GATES) c[i] = a[i] + b[i];
}

__global__ void k_fill_xh_global(const float* __restrict__ g, __bf16* __restrict__ xh) {
    ll n = (ll)B_ * D_;
    for (ll i = (ll)blockIdx.x * blockDim.x + threadIdx.x; i < n; i += (ll)gridDim.x * blockDim.x) {
        int b = (int)(i / D_), d = (int)(i % D_);
        xh[(ll)b * KCAT + E_ + d] = f2bf(g[i]);
    }
}

__global__ void k_init_tokens(int* tok) {
    int b = threadIdx.x;
    if (b < B_) tok[b] = 1;   // START_IDX
}

__global__ void k_init_state(const float* __restrict__ g,
                             const float* __restrict__ Wh, const float* __restrict__ Wm,
                             float* __restrict__ h, float* __restrict__ m, __bf16* __restrict__ xh) {
    ll n = (ll)B_ * H_;
    for (ll i = (ll)blockIdx.x * blockDim.x + threadIdx.x; i < n; i += (ll)gridDim.x * blockDim.x) {
        int b = (int)(i / H_), j = (int)(i % H_);
        const float* gr = g + (ll)b * D_;
        float ah = 0.f, am = 0.f;
        for (int d = 0; d < D_; ++d) {
            float gv = gr[d];
            ah += gv * Wh[(ll)d * H_ + j];
            am += gv * Wm[(ll)d * H_ + j];
        }
        float hv = tanhf(ah), mv = tanhf(am);
        h[i] = hv; m[i] = mv;
        xh[(ll)b * KCAT + XED + j] = f2bf(hv);
    }
}

// ---------------------------------------------------------------------------
// per-step kernels
// ---------------------------------------------------------------------------
__global__ void k_gather_x(const float* __restrict__ embed, const int* __restrict__ tok,
                           __bf16* __restrict__ xh) {
    int i = blockIdx.x * blockDim.x + threadIdx.x;
    if (i < B_ * E_) {
        int b = i / E_, e = i % E_;
        xh[(ll)b * KCAT + e] = f2bf(embed[(ll)tok[b] * E_ + e]);
    }
}

__global__ void k_lstm_cell(const float* __restrict__ gates,
                            float* __restrict__ h, float* __restrict__ m,
                            __bf16* __restrict__ xh) {
    int i = blockIdx.x * blockDim.x + threadIdx.x;
    if (i < B_ * H_) {
        int b = i / H_, j = i % H_;
        const float* gr = gates + (ll)b * GATES;
        float ig = sigm(gr[j]);
        float fg = sigm(gr[H_ + j]);
        float gg = tanhf(gr[2 * H_ + j]);
        float og = sigm(gr[3 * H_ + j]);
        float mn = fg * m[i] + ig * gg;
        float hn = og * tanhf(mn);
        m[i] = mn; h[i] = hn;
        xh[(ll)b * KCAT + XED + j] = f2bf(hn);
    }
}

__global__ __launch_bounds__(64) void k_attn_z(const float* __restrict__ sp,
                                               const float* __restrict__ gh,
                                               const float* __restrict__ wh,
                                               float* __restrict__ z) {
    __shared__ float red[64];
    int blk = blockIdx.x;              // b*L + l
    int b = blk / L_;
    const float* spr = sp + (ll)blk * K_;
    const float* ghr = gh + (ll)b * K_;
    float s = 0.f;
    for (int k = threadIdx.x; k < K_; k += 64)
        s += tanhf(spr[k] + ghr[k]) * wh[k];
    red[threadIdx.x] = s;
    __syncthreads();
    for (int off = 32; off > 0; off >>= 1) {
        if (threadIdx.x < off) red[threadIdx.x] += red[threadIdx.x + off];
        __syncthreads();
    }
    if (threadIdx.x == 0) z[blk] = red[0];
}

__global__ __launch_bounds__(256) void k_attn_ctx(const float* __restrict__ z,
                                                  const float* __restrict__ spatial,
                                                  const float* __restrict__ h,
                                                  __bf16* __restrict__ ch) {
    __shared__ float alpha[L_];
    __shared__ float stat[2];
    int b = blockIdx.x;
    const float* zr = z + b * L_;
    if (threadIdx.x == 0) {
        float mx = zr[0];
        for (int l = 1; l < L_; ++l) mx = fmaxf(mx, zr[l]);
        stat[0] = mx;
    }
    __syncthreads();
    if (threadIdx.x < L_) alpha[threadIdx.x] = expf(zr[threadIdx.x] - stat[0]);
    __syncthreads();
    if (threadIdx.x == 0) {
        float s = 0.f;
        for (int l = 0; l < L_; ++l) s += alpha[l];
        stat[1] = 1.0f / s;
    }
    __syncthreads();
    if (threadIdx.x < L_) alpha[threadIdx.x] *= stat[1];
    __syncthreads();
    const float* sb = spatial + (ll)b * L_ * D_;
    for (int d = threadIdx.x; d < D_; d += 256) {
        float acc = 0.f;
        for (int l = 0; l < L_; ++l) acc += alpha[l] * sb[(ll)l * D_ + d];
        ch[(ll)b * D_ + d] = f2bf(acc + h[(ll)b * D_ + d]);   // c + h_new, bf16 for logits GEMM
    }
}

__global__ __launch_bounds__(256) void k_argmax(const float* __restrict__ logits,
                                                int* __restrict__ tok,
                                                int* __restrict__ tok_out, int t) {
    __shared__ float vred[256];
    __shared__ int   ired[256];
    int b = blockIdx.x;
    const float* row = logits + ((ll)b * T_ + t) * V_;
    float best = -INFINITY; int bi = 0;
    for (int v = threadIdx.x; v < V_; v += 256) {
        float x = row[v];
        if (x > best) { best = x; bi = v; }     // strided: local indices increase
    }
    vred[threadIdx.x] = best; ired[threadIdx.x] = bi;
    __syncthreads();
    for (int off = 128; off > 0; off >>= 1) {
        if (threadIdx.x < off) {
            float ov = vred[threadIdx.x + off]; int oi = ired[threadIdx.x + off];
            if (ov > vred[threadIdx.x] || (ov == vred[threadIdx.x] && oi < ired[threadIdx.x])) {
                vred[threadIdx.x] = ov; ired[threadIdx.x] = oi;
            }
        }
        __syncthreads();
    }
    if (threadIdx.x == 0) { tok[b] = ired[0]; tok_out[(ll)b * T_ + t] = ired[0]; }
}

// ---------------------------------------------------------------------------
extern "C" void kernel_launch(void* const* d_in, const int* in_sizes, int n_in,
                              void* d_out, int out_size, void* d_ws, size_t ws_size,
                              hipStream_t stream) {
    (void)in_sizes; (void)n_in; (void)out_size; (void)ws_size;
    const float* spatial  = (const float*)d_in[0];
    const float* global_  = (const float*)d_in[1];
    const float* embed    = (const float*)d_in[2];
    const float* W_ih     = (const float*)d_in[3];
    const float* W_hh     = (const float*)d_in[4];
    const float* b_ih     = (const float*)d_in[5];
    const float* b_hh     = (const float*)d_in[6];
    const float* W_v      = (const float*)d_in[7];
    const float* W_g      = (const float*)d_in[8];
    const float* w_h      = (const float*)d_in[9];
    const float* W_p_w    = (const float*)d_in[10];
    const float* W_p_b    = (const float*)d_in[11];
    const float* W_init_h = (const float*)d_in[12];
    const float* W_init_m = (const float*)d_in[13];

    float* out = (float*)d_out;
    char* wsp = (char*)d_ws;
    auto alloc = [&](size_t bytes) -> char* {
        char* p = wsp; wsp += (bytes + 255) & ~(size_t)255; return p;
    };

    __bf16* Wcat   = (__bf16*)alloc((size_t)GATES * KCAT * 2);   // [8192,4608]
    __bf16* Wp     = (__bf16*)alloc((size_t)V_ * H_ * 2);        // [10000,2048]
    __bf16* WvT    = (__bf16*)alloc((size_t)K_ * D_ * 2);        // [512,2048]
    __bf16* WgT    = (__bf16*)alloc((size_t)K_ * H_ * 2);        // [512,2048]
    __bf16* spatbf = (__bf16*)alloc((size_t)B_ * L_ * D_ * 2);   // [6272,2048]
    float*  sproj  = (float*)alloc((size_t)B_ * L_ * K_ * 4);    // [6272,512]
    __bf16* xh     = (__bf16*)alloc((size_t)B_ * KCAT * 2);      // [128,4608] = [emb|glob|h]
    float*  gates  = (float*)alloc((size_t)B_ * GATES * 4);
    float*  hbuf   = (float*)alloc((size_t)B_ * H_ * 4);
    float*  mbuf   = (float*)alloc((size_t)B_ * H_ * 4);
    float*  gh     = (float*)alloc((size_t)B_ * K_ * 4);
    float*  zbuf   = (float*)alloc((size_t)B_ * L_ * 4);
    __bf16* ch     = (__bf16*)alloc((size_t)B_ * D_ * 2);
    float*  bcat   = (float*)alloc((size_t)GATES * 4);
    int*    tok    = (int*)alloc((size_t)B_ * 4);

    // one-time packing: bf16 weights stay L2-resident across the 20 steps
    k_build_wcat   <<<2048, 256, 0, stream>>>(W_ih, W_hh, Wcat);
    k_f32_to_bf16  <<<2048, 256, 0, stream>>>(W_p_w, Wp, (ll)V_ * H_);
    k_transpose_bf16<<<1024, 256, 0, stream>>>(W_v, WvT, D_, K_);
    k_transpose_bf16<<<1024, 256, 0, stream>>>(W_g, WgT, H_, K_);
    k_f32_to_bf16  <<<2048, 256, 0, stream>>>(spatial, spatbf, (ll)B_ * L_ * D_);
    k_fill_xh_global<<<1024, 256, 0, stream>>>(global_, xh);
    k_bias_cat     <<<GATES / 256, 256, 0, stream>>>(b_ih, b_hh, bcat);
    k_init_tokens  <<<1, 128, 0, stream>>>(tok);
    k_init_state   <<<1024, 256, 0, stream>>>(global_, W_init_h, W_init_m, hbuf, mbuf, xh);

    // spatial_proj = spatial @ W_v : [6272,2048] x [2048,512]
    gemm_bf16_wt<<<dim3(K_ / 64, (B_ * L_) / 128), 256, 0, stream>>>(
        spatbf, D_, WvT, D_, nullptr, sproj, K_, K_, D_);

    int* tok_out = (int*)(out + (ll)B_ * T_ * V_);

    for (int t = 0; t < T_; ++t) {
        k_gather_x<<<(B_ * E_) / 256, 256, 0, stream>>>(embed, tok, xh);
        // gates = [x|h] @ [W_ih|W_hh]^T + (b_ih + b_hh)
        gemm_bf16_wt<<<dim3(GATES / 64, 1), 256, 0, stream>>>(
            xh, KCAT, Wcat, KCAT, bcat, gates, GATES, GATES, KCAT);
        k_lstm_cell<<<(B_ * H_) / 256, 256, 0, stream>>>(gates, hbuf, mbuf, xh);
        // gh = h_new @ W_g  (A aliases the h-slot of xh)
        gemm_bf16_wt<<<dim3(K_ / 64, 1), 256, 0, stream>>>(
            xh + XED, KCAT, WgT, D_, nullptr, gh, K_, K_, H_);
        k_attn_z<<<B_ * L_, 64, 0, stream>>>(sproj, gh, w_h, zbuf);
        k_attn_ctx<<<B_, 256, 0, stream>>>(zbuf, spatial, hbuf, ch);
        // logits[:, t, :] = (c + h) @ W_p_w^T + W_p_b  -> write straight into d_out
        gemm_bf16_wt<<<dim3((V_ + 63) / 64, 1), 256, 0, stream>>>(
            ch, D_, Wp, D_, W_p_b, out + (ll)t * V_, (ll)T_ * V_, V_, D_);
        k_argmax<<<B_, 256, 0, stream>>>(out, tok, tok_out, t);
    }
}
